// Encoder_Block_8675833938629
// MI455X (gfx1250) — matrix operand order repaired and checked by reading, not compiled, not run on hardware
//
#include <hip/hip_runtime.h>
#include <hip/hip_bf16.h>

// ---------------------------------------------------------------------------
// Types for WMMA fragments (gfx1250, wave32)
// ---------------------------------------------------------------------------
typedef __attribute__((ext_vector_type(2)))  float   v2f;
typedef __attribute__((ext_vector_type(8)))  float   v8f;
typedef __bf16 bf16_t;
typedef __attribute__((ext_vector_type(8)))  __bf16  v8bf;
typedef __attribute__((ext_vector_type(16))) __bf16  v16bf;

#define B_  2
#define S_  2048
#define E_  1024
#define H_  16
#define D_  64
#define FE_ 4
#define EPS_ 1e-5f
#define SCALE_ 45.25483399593904f   /* sqrt(2048): faithful to reference quirk */

// ---------------------------------------------------------------------------
// Async global->LDS copy of 16 bytes per lane (gfx1250 ASYNCcnt path).
// LDS destination VGPR = low 32 bits of the flat shared address (addrspacecast
// AS3->flat keeps the LDS byte offset in addr[31:0], ISA 10.2).
// ---------------------------------------------------------------------------
__device__ __forceinline__ void async_cp16(void* lds_ptr, const void* gptr) {
    unsigned int       l = (unsigned int)(unsigned long long)lds_ptr;
    unsigned long long g = (unsigned long long)gptr;
    asm volatile("global_load_async_to_lds_b128 %0, %1, off"
                 :: "v"(l), "v"(g) : "memory");
}
__device__ __forceinline__ void async_wait0() {
    asm volatile("s_wait_asynccnt 0x0" ::: "memory");
}

// ---------------------------------------------------------------------------
// fp32 -> bf16 elementwise conversion (weights / activations feeding bf16 GEMM)
// ---------------------------------------------------------------------------
__global__ void cvt_bf16_kernel(const float* __restrict__ x, bf16_t* __restrict__ y) {
    int i = blockIdx.x * 256 + threadIdx.x;
    y[i] = (bf16_t)x[i];
}

// ---------------------------------------------------------------------------
// Per-head projection: C[M,64] = X[M,64] @ W^T   (W is [64,64] row-major)
// M = B*S*H = 65536.  fp32 WMMA 16x16x4 (attention path must stay fp32:
// softmax logits are scaled by sqrt(S)~45; bf16 here would flip argmaxes).
// One wave per 16x16 tile; grid exact so EXEC is all-ones at every WMMA.
// ---------------------------------------------------------------------------
__global__ void proj64_kernel(const float* __restrict__ X,
                              const float* __restrict__ W,
                              float* __restrict__ C) {
    int wid    = blockIdx.x * 8 + (threadIdx.x >> 5);   // global wave id
    int lane   = threadIdx.x & 31;
    int hi     = lane >> 4;        // lane half
    int lr     = lane & 15;
    int tile_m = wid >> 2;         // 4096 m-tiles
    int tile_n = wid & 3;          // 4 n-tiles (N = 64)
    int m0 = tile_m * 16, n0 = tile_n * 16;

    const float* xrow = X + (size_t)(m0 + lr) * 64;  // A row, lane holds M=lr
    const float* wrow = W + (size_t)(n0 + lr) * 64;  // B[e][n] = W[n][e]

    v8f c = {};
    #pragma unroll
    for (int k = 0; k < 64; k += 4) {
        int ka = k + 2 * hi;                    // f32 A/B frag: K = 2*(lane/16)+i
        v2f a = *(const v2f*)(xrow + ka);
        v2f b = *(const v2f*)(wrow + ka);
        c = __builtin_amdgcn_wmma_f32_16x16x4_f32(false, a, false, b,
                                                  (short)0, c, false, false);
    }
    // C/D layout: VGPR j -> row m0 + j + 8*hi, col n0 + lr
    float* crow = C + (size_t)m0 * 64 + n0 + lr;
    #pragma unroll
    for (int j = 0; j < 8; ++j)
        crow[(size_t)(j + 8 * hi) * 64] = c[j];
}

// ---------------------------------------------------------------------------
// Flash attention per (b,h).  grid = (S/64, B*H), block = 128 (4 waves).
// Each wave owns 16 query rows; Q fragments resident in VGPRs.
// Per 16-key tile: 16x f32 WMMA (Q@K^T) -> LDS (17-padded, conflict-free)
// -> online softmax (lanes 0..15, one row each) -> rescale O -> 16x f32 WMMA
// (P@V).  O normalized by running row-sum at the end.
// ---------------------------------------------------------------------------
__global__ void flash_kernel(const float* __restrict__ q,
                             const float* __restrict__ k,
                             const float* __restrict__ v,
                             float* __restrict__ o) {
    int bh = blockIdx.y;
    int b  = bh >> 4;           // H_ = 16
    int h  = bh & 15;
    int wv   = threadIdx.x >> 5;
    int lane = threadIdx.x & 31;
    int hi   = lane >> 4;
    int lr   = lane & 15;
    int qrow0 = blockIdx.x * 64 + wv * 16;

    const float* qb = q + (size_t)b * S_ * E_ + h * D_;   // [s][*] stride E_
    const float* kb = k + (size_t)b * S_ * E_ + h * D_;
    const float* vb = v + (size_t)b * S_ * E_ + h * D_;

    // Load Q fragments once: A frag t covers K = 4t + 2*hi + {0,1}
    v2f aq[16];
    {
        const float* qr = qb + (size_t)(qrow0 + lr) * E_;
        #pragma unroll
        for (int t = 0; t < 16; ++t)
            aq[t] = *(const v2f*)(qr + 4 * t + 2 * hi);
    }

    v8f acc[4] = {};            // O accumulator: 16 rows x 64 cols (4 n-tiles)
    float m_run = -1e30f;       // per-row running max (valid on lanes 0..15)
    float l_run = 0.f;          // per-row running sum

    __shared__ float Tls[4][16 * 17];   // 17-padded score/prob tile per wave
    __shared__ float alpha_s[4][16];
    __shared__ float lsum_s[4][16];

    for (int jt = 0; jt < S_ / 16; ++jt) {
        // ---- S = Q @ K_tile^T (fp32 WMMA) ----
        v8f c = {};
        const float* kr = kb + (size_t)(jt * 16 + lr) * E_;  // B[e][n]=K[n][e]
        #pragma unroll
        for (int t = 0; t < 16; ++t) {
            v2f bk = *(const v2f*)(kr + 4 * t + 2 * hi);
            c = __builtin_amdgcn_wmma_f32_16x16x4_f32(false, aq[t], false, bk,
                                                      (short)0, c, false, false);
        }
        #pragma unroll
        for (int j = 0; j < 8; ++j)
            Tls[wv][(j + 8 * hi) * 17 + lr] = c[j] * SCALE_;
        __syncthreads();

        // ---- online softmax: lane r handles row r ----
        if (lane < 16) {
            float* row = &Tls[wv][lane * 17];
            float mx = m_run;
            #pragma unroll
            for (int jj = 0; jj < 16; ++jj) mx = fmaxf(mx, row[jj]);
            float al = __expf(m_run - mx);
            float s  = 0.f;
            #pragma unroll
            for (int jj = 0; jj < 16; ++jj) {
                float p = __expf(row[jj] - mx);
                row[jj] = p;
                s += p;
            }
            l_run = l_run * al + s;
            m_run = mx;
            alpha_s[wv][lane] = al;
        }
        __syncthreads();

        // ---- rescale accumulators by alpha(row) ----
        float alj[8];
        #pragma unroll
        for (int j = 0; j < 8; ++j) alj[j] = alpha_s[wv][j + 8 * hi];
        #pragma unroll
        for (int nt = 0; nt < 4; ++nt)
            #pragma unroll
            for (int j = 0; j < 8; ++j) acc[nt][j] *= alj[j];

        // ---- O += P @ V_tile (fp32 WMMA); P read from LDS in A layout ----
        #pragma unroll
        for (int t2 = 0; t2 < 4; ++t2) {
            int kk = 4 * t2 + 2 * hi;                  // key index within tile
            v2f ap;
            ap.x = Tls[wv][lr * 17 + kk];
            ap.y = Tls[wv][lr * 17 + kk + 1];
            const float* vr0 = vb + (size_t)(jt * 16 + kk) * E_;
            const float* vr1 = vr0 + E_;
            #pragma unroll
            for (int nt = 0; nt < 4; ++nt) {
                v2f bv;
                bv.x = vr0[nt * 16 + lr];
                bv.y = vr1[nt * 16 + lr];
                acc[nt] = __builtin_amdgcn_wmma_f32_16x16x4_f32(
                    false, ap, false, bv, (short)0, acc[nt], false, false);
            }
        }
        __syncthreads();   // protect Tls before next iteration overwrites it
    }

    if (lane < 16) lsum_s[wv][lane] = l_run;
    __syncthreads();

    float* ob = o + (size_t)b * S_ * E_ + h * D_;
    #pragma unroll
    for (int j = 0; j < 8; ++j) {
        float inv = 1.f / lsum_s[wv][j + 8 * hi];
        float* op = ob + (size_t)(qrow0 + j + 8 * hi) * E_;
        #pragma unroll
        for (int nt = 0; nt < 4; ++nt)
            op[nt * 16 + lr] = acc[nt][j] * inv;
    }
}

// ---------------------------------------------------------------------------
// bf16 GEMM with fp32 accumulate: C[M,N] = A[M,K] @ W^T (+bias)(+resid)(relu)
// W is [N,K] row-major.  Block tile 64x128, 8 waves of 32x32 (2x2 c-frags).
// Double-buffered LDS pipeline fed by GLOBAL_LOAD_ASYNC_TO_LDS_B128:
// stage k-step i+1 while WMMAs consume k-step i; s_wait_asynccnt + barrier
// publishes the tile block-wide.  LDS rows padded to 80B (20 dwords,
// gcd(20,64)=4 -> fragment ds_load_b128 row-starts hit distinct banks).
// ---------------------------------------------------------------------------
__global__ void gemm_bf16_kernel(const bf16_t* __restrict__ A,
                                 const bf16_t* __restrict__ W,
                                 const float* __restrict__ bias,
                                 const float* __restrict__ resid,
                                 float* __restrict__ outf,
                                 bf16_t* __restrict__ outb,
                                 int M, int N, int K, int do_relu) {
    __shared__ bf16_t smA[2][64 * 40];     // 64 rows x 32 cols, stride 40
    __shared__ bf16_t smB[2][128 * 40];    // 128 rows x 32 cols, stride 40

    int tid  = threadIdx.x;
    int wvid = tid >> 5;
    int lane = tid & 31;
    int hi   = lane >> 4;
    int lr   = lane & 15;
    int nblk = N >> 7;                       // N / 128
    int bm = blockIdx.x / nblk, bn = blockIdx.x % nblk;
    int wm = wvid >> 2, wn = wvid & 3;
    int m0 = bm * 64;
    int n0 = bn * 128;

    // Staging assignment: 16B chunks. A tile = 256 chunks (1/thread),
    // B tile = 512 chunks (2/thread).  chunk c -> row c>>2, seg c&3.
    int srow = tid >> 2, sseg = tid & 3;
    const bf16_t* ga  = A + (size_t)(m0 + srow) * K + sseg * 8;
    const bf16_t* gb0 = W + (size_t)(n0 + srow) * K + sseg * 8;
    const bf16_t* gb1 = W + (size_t)(n0 + 64 + srow) * K + sseg * 8;
    int lofs = srow * 40 + sseg * 8;         // element offset in padded LDS row

    v8f c[2][2] = {};
    const int nk = K >> 5;

    // prologue: stage k-step 0 into buffer 0
    async_cp16(&smA[0][lofs], ga);
    async_cp16(&smB[0][lofs], gb0);
    async_cp16(&smB[0][64 * 40 + lofs], gb1);
    async_wait0();
    __syncthreads();

    for (int ks = 0; ks < nk; ++ks) {
        int buf = ks & 1;
        if (ks + 1 < nk) {                   // stage next k-step into other buf
            int ko = (ks + 1) * 32;
            async_cp16(&smA[buf ^ 1][lofs], ga + ko);
            async_cp16(&smB[buf ^ 1][lofs], gb0 + ko);
            async_cp16(&smB[buf ^ 1][64 * 40 + lofs], gb1 + ko);
        }

        // ---- fragments from LDS ----
        const bf16_t* pa0 = &smA[buf][(wm * 32 + lr) * 40];
        const bf16_t* pa1 = &smA[buf][(wm * 32 + 16 + lr) * 40];
        const bf16_t* pb0 = &smB[buf][(wn * 32 + lr) * 40];
        const bf16_t* pb1 = &smB[buf][(wn * 32 + 16 + lr) * 40];

        // A frag: elems 0..7 at K=8*hi, elems 8..15 at K=8*hi+16
        v8bf a0lo = *(const v8bf*)(pa0 + 8 * hi);
        v8bf a0hi = *(const v8bf*)(pa0 + 8 * hi + 16);
        v8bf a1lo = *(const v8bf*)(pa1 + 8 * hi);
        v8bf a1hi = *(const v8bf*)(pa1 + 8 * hi + 16);
        v16bf a0 = __builtin_shufflevector(a0lo, a0hi,
                     0,1,2,3,4,5,6,7,8,9,10,11,12,13,14,15);
        v16bf a1 = __builtin_shufflevector(a1lo, a1hi,
                     0,1,2,3,4,5,6,7,8,9,10,11,12,13,14,15);
        // B frag: elems 0..7 at K=8*hi, elems 8..15 at K=8*hi+16 (16B-aligned)
        v8bf b0lo = *(const v8bf*)(pb0 + 8 * hi);
        v8bf b0hi = *(const v8bf*)(pb0 + 8 * hi + 16);
        v8bf b1lo = *(const v8bf*)(pb1 + 8 * hi);
        v8bf b1hi = *(const v8bf*)(pb1 + 8 * hi + 16);
        v16bf b0 = __builtin_shufflevector(b0lo, b0hi,
                     0,1,2,3,4,5,6,7,8,9,10,11,12,13,14,15);
        v16bf b1 = __builtin_shufflevector(b1lo, b1hi,
                     0,1,2,3,4,5,6,7,8,9,10,11,12,13,14,15);

        c[0][0] = __builtin_amdgcn_wmma_f32_16x16x32_bf16(false, a0, false, b0,
                                                          (short)0, c[0][0], false, false);
        c[0][1] = __builtin_amdgcn_wmma_f32_16x16x32_bf16(false, a0, false, b1,
                                                          (short)0, c[0][1], false, false);
        c[1][0] = __builtin_amdgcn_wmma_f32_16x16x32_bf16(false, a1, false, b0,
                                                          (short)0, c[1][0], false, false);
        c[1][1] = __builtin_amdgcn_wmma_f32_16x16x32_bf16(false, a1, false, b1,
                                                          (short)0, c[1][1], false, false);

        async_wait0();      // our staged chunks for buf^1 are in LDS
        __syncthreads();    // everyone's chunks are in LDS / buf reads done
    }

    // ---- epilogue: bias (+resid)(relu), fp32 and/or bf16 stores ----
    #pragma unroll
    for (int mi = 0; mi < 2; ++mi) {
        #pragma unroll
        for (int ni = 0; ni < 2; ++ni) {
            int col = n0 + wn * 32 + ni * 16 + lr;
            float bsv = bias[col];
            #pragma unroll
            for (int j = 0; j < 8; ++j) {
                int row = m0 + wm * 32 + mi * 16 + j + 8 * hi;
                size_t idx = (size_t)row * N + col;
                float val = c[mi][ni][j] + bsv;
                if (resid) val += resid[idx];
                if (do_relu) val = fmaxf(val, 0.f);
                if (outf) outf[idx] = val;
                if (outb) outb[idx] = (bf16_t)val;
            }
        }
    }
}

// ---------------------------------------------------------------------------
// LayerNorm over last dim (1024). One 256-thread block per row, 4 elems/thread.
// Optionally emits a bf16 copy (input to the next bf16 GEMM).
// ---------------------------------------------------------------------------
__global__ void ln_kernel(const float* __restrict__ x,
                          const float* __restrict__ g,
                          const float* __restrict__ bta,
                          float* __restrict__ outf,
                          bf16_t* __restrict__ outb) {
    __shared__ float r1[256], r2[256];
    int row = blockIdx.x, tid = threadIdx.x;
    const float* xr = x + (size_t)row * E_;
    float vals[4], s = 0.f, s2 = 0.f;
    #pragma unroll
    for (int i = 0; i < 4; ++i) {
        float t = xr[tid + i * 256];
        vals[i] = t;
        s += t;
        s2 += t * t;
    }
    r1[tid] = s; r2[tid] = s2;
    __syncthreads();
    for (int off = 128; off; off >>= 1) {
        if (tid < off) { r1[tid] += r1[tid + off]; r2[tid] += r2[tid + off]; }
        __syncthreads();
    }
    float mu  = r1[0] * (1.f / E_);
    float var = r2[0] * (1.f / E_) - mu * mu;
    float rstd = rsqrtf(var + EPS_);
    #pragma unroll
    for (int i = 0; i < 4; ++i) {
        int cidx = tid + i * 256;
        float y = (vals[i] - mu) * rstd * g[cidx] + bta[cidx];
        outf[(size_t)row * E_ + cidx] = y;
        if (outb) outb[(size_t)row * E_ + cidx] = (bf16_t)y;
    }
}

// ---------------------------------------------------------------------------
// Host launcher
// ---------------------------------------------------------------------------
extern "C" void kernel_launch(void* const* d_in, const int* in_sizes, int n_in,
                              void* d_out, int out_size, void* d_ws, size_t ws_size,
                              hipStream_t stream) {
    (void)in_sizes; (void)n_in; (void)out_size; (void)ws_size;
    const float* queries = (const float*)d_in[0];
    const float* keys    = (const float*)d_in[1];
    const float* values  = (const float*)d_in[2];
    /* d_in[3] = mask: ignored, faithful to reference */
    const float* Wq  = (const float*)d_in[4];
    const float* Wk  = (const float*)d_in[5];
    const float* Wv  = (const float*)d_in[6];
    const float* Wfc = (const float*)d_in[7];
    const float* bfc = (const float*)d_in[8];
    const float* W1  = (const float*)d_in[9];
    const float* b1  = (const float*)d_in[10];
    const float* W2  = (const float*)d_in[11];
    const float* b2  = (const float*)d_in[12];
    const float* ln1g = (const float*)d_in[13];
    const float* ln1b = (const float*)d_in[14];
    const float* ln2g = (const float*)d_in[15];
    const float* ln2b = (const float*)d_in[16];
    float* out = (float*)d_out;

    const size_t MN  = (size_t)B_ * S_;       // 4096 rows
    const size_t ACT = MN * E_;               // 4M elems

    char* ws = (char*)d_ws;
    size_t off = 0;
    auto take = [&](size_t bytes) -> void* {
        void* p = ws + off;
        off += (bytes + 255) & ~(size_t)255;
        return p;
    };
    float*  qb     = (float*)take(ACT * 4);
    float*  kbuf   = (float*)take(ACT * 4);
    float*  vbuf   = (float*)take(ACT * 4);
    float*  att    = (float*)take(ACT * 4);
    bf16_t* attb   = (bf16_t*)take(ACT * 2);
    float*  preln1 = (float*)take(ACT * 4);
    float*  hbuf   = (float*)take(ACT * 4);
    bf16_t* hb     = (bf16_t*)take(ACT * 2);
    bf16_t* ub     = (bf16_t*)take(MN * FE_ * E_ * 2);
    float*  preln2 = (float*)take(ACT * 4);
    bf16_t* Wfcb   = (bf16_t*)take((size_t)E_ * E_ * 2);
    bf16_t* W1b    = (bf16_t*)take((size_t)FE_ * E_ * E_ * 2);
    bf16_t* W2b    = (bf16_t*)take((size_t)FE_ * E_ * E_ * 2);

    // Weight downcasts (bf16 GEMM path; LN bounds the relative error)
    cvt_bf16_kernel<<<(E_ * E_) / 256, 256, 0, stream>>>(Wfc, Wfcb);
    cvt_bf16_kernel<<<(FE_ * E_ * E_) / 256, 256, 0, stream>>>(W1, W1b);
    cvt_bf16_kernel<<<(FE_ * E_ * E_) / 256, 256, 0, stream>>>(W2, W2b);

    // Per-head Q/K/V projections (fp32 WMMA), M = B*S*H = 65536
    proj64_kernel<<<2048, 256, 0, stream>>>(queries, Wq, qb);
    proj64_kernel<<<2048, 256, 0, stream>>>(keys,    Wk, kbuf);
    proj64_kernel<<<2048, 256, 0, stream>>>(values,  Wv, vbuf);

    // Flash attention with the reference's sqrt(S) scale quirk
    flash_kernel<<<dim3(S_ / 64, B_ * H_), 128, 0, stream>>>(qb, kbuf, vbuf, att);

    // att -> bf16 for the fc GEMM
    cvt_bf16_kernel<<<(int)(ACT / 256), 256, 0, stream>>>(att, attb);

    // preln1 = att @ Wfc^T + bfc + queries
    gemm_bf16_kernel<<<(MN / 64) * (E_ / 128), 256, 0, stream>>>(
        attb, Wfcb, bfc, queries, preln1, nullptr, (int)MN, E_, E_, 0);

    // h = LN1(preln1), also bf16 copy
    ln_kernel<<<(int)MN, 256, 0, stream>>>(preln1, ln1g, ln1b, hbuf, hb);

    // u = relu(h @ W1^T + b1), bf16 out only
    gemm_bf16_kernel<<<(MN / 64) * ((FE_ * E_) / 128), 256, 0, stream>>>(
        hb, W1b, b1, nullptr, nullptr, ub, (int)MN, FE_ * E_, E_, 1);

    // preln2 = u @ W2^T + b2 + h
    gemm_bf16_kernel<<<(MN / 64) * (E_ / 128), 256, 0, stream>>>(
        ub, W2b, b2, hbuf, preln2, nullptr, (int)MN, E_, FE_ * E_, 0);

    // out = LN2(preln2)
    ln_kernel<<<(int)MN, 256, 0, stream>>>(preln2, ln2g, ln2b, out, nullptr);
}
